// DecoderRNN_71064528880215
// MI455X (gfx1250) — compile-verified, matrix operand order checked by reading, not run on hardware
//
#include <hip/hip_runtime.h>
#include <hip/hip_bf16.h>

// ---------------------------------------------------------------------------
// DecoderRNN: 2-layer GRU (B=64, T-1=255 steps, D=1024, H=2048) + 3-layer MLP.
//  * bf16 WMMA (v_wmma_f32_16x16x32_bf16), fp32 accumulate.
//  * Recurrent weights (~88MB bf16) stay L2-resident on MI455X (192MB L2).
//  * GRU step: A-tile (activations) staged in LDS in 16KB chunks via the
//    gfx1250 async global->LDS path, shared by all 4 waves of the block;
//    weights stream from L2. Gate math fused in the WMMA epilogue.
//  * MLP head: WMMA GEMM, 2 M-tiles x 4 N-tiles per wave (halves weight
//    traffic), bias+ReLU fused.
// Requires ws_size >= ~320 MB.
// ---------------------------------------------------------------------------

typedef __attribute__((ext_vector_type(16))) __bf16 v16bf;
typedef __attribute__((ext_vector_type(8)))  __bf16 v8bf;
typedef __attribute__((ext_vector_type(8)))  float  v8f;
typedef __attribute__((ext_vector_type(4)))  int    v4i;

constexpr int B_  = 64;
constexpr int T_  = 256;
constexpr int TS_ = 255;      // T-1 timesteps
constexpr int D_  = 1024;
constexpr int H_  = 2048;
constexpr int KC_ = 512;      // K-chunk staged in LDS: 16 x 512 bf16 = 16KB
constexpr long MROWS = (long)B_ * TS_;   // 16320 rows for the MLP head

#if defined(__has_builtin)
#if __has_builtin(__builtin_amdgcn_global_load_async_to_lds_b128)
#define USE_ASYNC_LDS 1
#endif
#endif
#ifndef USE_ASYNC_LDS
#define USE_ASYNC_LDS 0
#endif

__device__ __forceinline__ float sigmoidf_(float x) {
  return 1.0f / (1.0f + __expf(-x));
}

// A-fragment (16x32, bf16): lane<16 -> row=lane, K in {0..7, 16..23};
// lane>=16 -> row=lane-16, K in {8..15, 24..31}.  khalf = 0 or 8.
// B-fragment symmetric (N takes role of M); gh = h @ W^T => B[k][n] = W[n][k],
// contiguous along k in row n of W -> same loader, two b128 loads per lane.
__device__ __forceinline__ v16bf ldfrag(const __bf16* __restrict__ p, int khalf) {
  v8bf lo = *(const v8bf*)(p + khalf);
  v8bf hi = *(const v8bf*)(p + 16 + khalf);
  v16bf r;
#pragma unroll
  for (int i = 0; i < 8; ++i) { r[i] = lo[i]; r[8 + i] = hi[i]; }
  return r;
}

__device__ __forceinline__ v8f wmma_bf16(v16bf a, v16bf b, v8f c) {
  return __builtin_amdgcn_wmma_f32_16x16x32_bf16(
      /*neg_a=*/false, a, /*neg_b=*/false, b,
      /*c_mod=*/(short)0, c, /*reuse_a=*/false, /*reuse_b=*/false);
}

// copy 16 bytes global -> LDS (async DMA if the toolchain exposes it)
__device__ __forceinline__ void cp_g2l_16(const __bf16* g, __bf16* l) {
#if USE_ASYNC_LDS
  __builtin_amdgcn_global_load_async_to_lds_b128(
      (__attribute__((address_space(1))) v4i*)(uintptr_t)g,
      (__attribute__((address_space(3))) v4i*)(v4i*)l,
      0, 0);
#else
  *(v8bf*)l = *(const v8bf*)g;
#endif
}

__device__ __forceinline__ void cp_g2l_commit() {
#if USE_ASYNC_LDS
#if defined(__has_builtin) && __has_builtin(__builtin_amdgcn_s_wait_asynccnt)
  __builtin_amdgcn_s_wait_asynccnt(0);
#else
  asm volatile("s_wait_asynccnt 0x0" ::: "memory");
#endif
#endif
  __syncthreads();
}

// ---------------------------------------------------------------------------
// fp32 -> bf16 conversion (weights, inputs)
// ---------------------------------------------------------------------------
__global__ void cvt_bf16(const float* __restrict__ s, __bf16* __restrict__ d, long n) {
  long i = (long)blockIdx.x * blockDim.x + threadIdx.x;
  long stride = (long)gridDim.x * blockDim.x;
  for (; i < n; i += stride) d[i] = (__bf16)s[i];
}

// hidden (2,B,H) fp32 -> ping buffer 0 (fp32 + bf16 copies)
__global__ void init_hidden(const float* __restrict__ hid,
                            float* __restrict__ h0f, __bf16* __restrict__ h0b,
                            float* __restrict__ h1f, __bf16* __restrict__ h1b) {
  long i = (long)blockIdx.x * blockDim.x + threadIdx.x;
  const long BH = (long)B_ * H_;
  if (i >= 2 * BH) return;
  float v = hid[i];
  if (i < BH) { h0f[i] = v; h0b[i] = (__bf16)v; }
  else        { h1f[i - BH] = v; h1b[i - BH] = (__bf16)v; }
}

__global__ void copy_hidden_out(const float* __restrict__ h0f,
                                const float* __restrict__ h1f,
                                float* __restrict__ out) {
  long i = (long)blockIdx.x * blockDim.x + threadIdx.x;
  const long BH = (long)B_ * H_;
  if (i >= 2 * BH) return;
  out[i] = (i < BH) ? h0f[i] : h1f[i - BH];
}

// ---------------------------------------------------------------------------
// Fused GRU step (one layer, one timestep).
// grid = (B/16, H/64), block = (32,4): wave owns one 16x16 tile of h_new.
// The 16xK activation tile (x segment, then h segment) is staged into LDS in
// KC_-wide chunks and shared by the 4 waves; gate pre-activations use 4 WMMA
// accumulators (r, z fully combined; n split into input/hidden because r
// multiplies only the hidden part).
// ---------------------------------------------------------------------------
__global__ __launch_bounds__(128)
void gru_step(const __bf16* __restrict__ xbf, long xRowStride, int xDim,
              const __bf16* __restrict__ Wi,   // (3H, xDim) row-major bf16
              const __bf16* __restrict__ Wh,   // (3H, H)    row-major bf16
              const float*  __restrict__ bih, const float* __restrict__ bhh,
              const float*  __restrict__ hprevF, const __bf16* __restrict__ hprevB,
              float* __restrict__ hnewF, __bf16* __restrict__ hnewB,
              __bf16* __restrict__ ys, int t) {
  __shared__ __align__(16) __bf16 sA[16 * KC_];

  const int lane  = threadIdx.x;                    // 0..31
  const int wave  = threadIdx.y;                    // 0..3
  const int tid   = wave * 32 + lane;
  const int m0    = blockIdx.x * 16;                // batch tile
  const int n0    = (blockIdx.y * 4 + wave) * 16;   // hidden tile
  const int lrow  = lane & 15;
  const int khalf = (lane >> 4) << 3;               // 0 or 8
  const int ncol  = n0 + lrow;

  v8f cr = {}, cz = {}, cni = {}, cnh = {};

#pragma unroll
  for (int seg = 0; seg < 2; ++seg) {
    const bool isH       = (seg == 1);
    const __bf16* src    = isH ? hprevB : xbf;
    const long   astride = isH ? (long)H_ : xRowStride;
    const int    Kseg    = isH ? H_ : xDim;
    const int    wstr    = isH ? H_ : xDim;
    const __bf16* Wseg   = isH ? Wh : Wi;
    const __bf16* w_r = Wseg + (long)(0 * H_ + ncol) * wstr;
    const __bf16* w_z = Wseg + (long)(1 * H_ + ncol) * wstr;
    const __bf16* w_n = Wseg + (long)(2 * H_ + ncol) * wstr;

    for (int kb = 0; kb < Kseg; kb += KC_) {
      // ---- cooperative stage of 16 x KC_ A-chunk into LDS ----
      for (int i = tid; i < 16 * (KC_ / 8); i += 128) {
        const int row = i >> 6;             // KC_/8 = 64 vectors per row
        const int col = (i & 63) << 3;
        cp_g2l_16(src + (long)(m0 + row) * astride + kb + col,
                  &sA[row * KC_ + col]);
      }
      cp_g2l_commit();

      const __bf16* aL = &sA[lrow * KC_];
      for (int kk = 0; kk < KC_; kk += 32) {
        __builtin_prefetch(w_r + kb + kk + 256, 0, 3);
        v16bf a  = ldfrag(aL + kk, khalf);
        v16bf br = ldfrag(w_r + kb + kk, khalf);
        v16bf bz = ldfrag(w_z + kb + kk, khalf);
        v16bf bn = ldfrag(w_n + kb + kk, khalf);
        cr = wmma_bf16(a, br, cr);
        cz = wmma_bf16(a, bz, cz);
        if (isH) cnh = wmma_bf16(a, bn, cnh);
        else     cni = wmma_bf16(a, bn, cni);
      }
      __syncthreads();   // protect sA before next chunk overwrites it
    }
  }

  // epilogue: gate math.  C/D layout: VGPR i, lane<16 -> M=i, lane>=16 -> M=8+i,
  // N = lane&15 -> each lane handles 8 rows of a single column.
  const float bsr = bih[ncol]          + bhh[ncol];
  const float bsz = bih[H_ + ncol]     + bhh[H_ + ncol];
  const float bin = bih[2 * H_ + ncol];
  const float bhn = bhh[2 * H_ + ncol];
  const int mbase = m0 + khalf;

#pragma unroll
  for (int i = 0; i < 8; ++i) {
    const int m  = mbase + i;
    const long o = (long)m * H_ + ncol;
    float hp = hprevF[o];
    float rg = sigmoidf_(cr[i] + bsr);
    float zg = sigmoidf_(cz[i] + bsz);
    float ng = tanhf(cni[i] + bin + rg * (cnh[i] + bhn));
    float hv = (1.0f - zg) * ng + zg * hp;
    hnewF[o] = hv;
    hnewB[o] = (__bf16)hv;
    if (ys) ys[((long)m * TS_ + t) * (long)H_ + ncol] = (__bf16)hv;  // (B,T-1,H)
  }
}

// ---------------------------------------------------------------------------
// WMMA GEMM with fused bias (+ optional ReLU):  out = act(A @ W^T + bias)
// A: (M,K) bf16 row-major; W: (N,K) bf16 row-major.
// grid = (M/32, N/256), block = (32,4); each wave does 2 M-tiles x 4 N-tiles
// (B fragments reused across both M-tiles, A fragments across the 4 N-tiles).
// ---------------------------------------------------------------------------
__global__ __launch_bounds__(128)
void gemm_bias_act(const __bf16* __restrict__ A, const __bf16* __restrict__ W,
                   const float* __restrict__ bias,
                   float* __restrict__ outF, __bf16* __restrict__ outB,
                   int K, int N, int act) {
  const int lane  = threadIdx.x;
  const int wave  = threadIdx.y;
  const int m0    = blockIdx.x * 32;
  const int nbase = blockIdx.y * 256 + wave * 64;
  const int lrow  = lane & 15;
  const int khalf = (lane >> 4) << 3;

  const __bf16* arow0 = A + (long)(m0 + lrow) * K;
  const __bf16* arow1 = A + (long)(m0 + 16 + lrow) * K;
  const __bf16* wrow[4];
#pragma unroll
  for (int j = 0; j < 4; ++j)
    wrow[j] = W + (long)(nbase + j * 16 + lrow) * K;

  v8f acc[2][4] = {};
  for (int k = 0; k < K; k += 32) {
    __builtin_prefetch(wrow[0] + k + 256, 0, 3);
    v16bf a0 = ldfrag(arow0 + k, khalf);
    v16bf a1 = ldfrag(arow1 + k, khalf);
#pragma unroll
    for (int j = 0; j < 4; ++j) {
      v16bf b = ldfrag(wrow[j] + k, khalf);
      acc[0][j] = wmma_bf16(a0, b, acc[0][j]);
      acc[1][j] = wmma_bf16(a1, b, acc[1][j]);
    }
  }

#pragma unroll
  for (int mi = 0; mi < 2; ++mi) {
#pragma unroll
    for (int j = 0; j < 4; ++j) {
      const int n = nbase + j * 16 + lrow;
      const float bv = bias[n];
#pragma unroll
      for (int i = 0; i < 8; ++i) {
        const int m = m0 + mi * 16 + khalf + i;
        float v = acc[mi][j][i] + bv;
        if (act) v = fmaxf(v, 0.0f);
        const long o = (long)m * N + n;
        if (outF) outF[o] = v;
        if (outB) outB[o] = (__bf16)v;
      }
    }
  }
}

// ---------------------------------------------------------------------------
// Host-side orchestration
// ---------------------------------------------------------------------------
extern "C" void kernel_launch(void* const* d_in, const int* in_sizes, int n_in,
                              void* d_out, int out_size, void* d_ws, size_t ws_size,
                              hipStream_t stream) {
  const float* target = (const float*)d_in[0];   // (B,T,D)
  const float* hidden = (const float*)d_in[1];   // (2,B,H)
  const float* Wih0   = (const float*)d_in[2];
  const float* Whh0   = (const float*)d_in[3];
  const float* bih0   = (const float*)d_in[4];
  const float* bhh0   = (const float*)d_in[5];
  const float* Wih1   = (const float*)d_in[6];
  const float* Whh1   = (const float*)d_in[7];
  const float* bih1   = (const float*)d_in[8];
  const float* bhh1   = (const float*)d_in[9];
  const float* W1     = (const float*)d_in[10];
  const float* b1     = (const float*)d_in[11];
  const float* W2     = (const float*)d_in[12];
  const float* b2     = (const float*)d_in[13];
  const float* W3     = (const float*)d_in[14];
  const float* b3     = (const float*)d_in[15];

  // workspace carving (element counts; all sizes even -> alignment preserved)
  constexpr long SZ_Wih0 = 3L * H_ * D_;       // 6144 x 1024
  constexpr long SZ_WhhX = 3L * H_ * H_;       // 6144 x 2048 (Whh0, Wih1, Whh1)
  constexpr long SZ_W1   = (long)H_ * H_;
  constexpr long SZ_W2   = (long)D_ * H_;
  constexpr long SZ_W3   = (long)D_ * D_;
  constexpr long SZ_X    = (long)B_ * T_ * D_;
  constexpr long SZ_YS   = MROWS * H_;
  constexpr long SZ_M2   = MROWS * D_;
  constexpr long SZ_HB   = (long)B_ * H_;

  __bf16* p = (__bf16*)d_ws;
  __bf16* wih0b = p; p += SZ_Wih0;
  __bf16* whh0b = p; p += SZ_WhhX;
  __bf16* wih1b = p; p += SZ_WhhX;
  __bf16* whh1b = p; p += SZ_WhhX;
  __bf16* w1b   = p; p += SZ_W1;
  __bf16* w2b   = p; p += SZ_W2;
  __bf16* w3b   = p; p += SZ_W3;
  __bf16* xb    = p; p += SZ_X;
  __bf16* ysb   = p; p += SZ_YS;
  __bf16* m1b   = p; p += SZ_YS;
  __bf16* m2b   = p; p += SZ_M2;
  __bf16* h0b[2]; h0b[0] = p; p += SZ_HB; h0b[1] = p; p += SZ_HB;
  __bf16* h1b[2]; h1b[0] = p; p += SZ_HB; h1b[1] = p; p += SZ_HB;
  float* f = (float*)p;
  float* h0f[2]; h0f[0] = f; f += SZ_HB; h0f[1] = f; f += SZ_HB;
  float* h1f[2]; h1f[0] = f; f += SZ_HB; h1f[1] = f; f += SZ_HB;

  // ---- convert weights / inputs to bf16 ----
  auto cvt = [&](const float* s, __bf16* d, long n) {
    int blocks = (int)((n + 1023) / 1024);
    if (blocks > 4096) blocks = 4096;
    cvt_bf16<<<blocks, 1024, 0, stream>>>(s, d, n);
  };
  cvt(Wih0, wih0b, SZ_Wih0);
  cvt(Whh0, whh0b, SZ_WhhX);
  cvt(Wih1, wih1b, SZ_WhhX);
  cvt(Whh1, whh1b, SZ_WhhX);
  cvt(W1,   w1b,   SZ_W1);
  cvt(W2,   w2b,   SZ_W2);
  cvt(W3,   w3b,   SZ_W3);
  cvt(target, xb,  SZ_X);

  init_hidden<<<(int)((2 * SZ_HB + 255) / 256), 256, 0, stream>>>(
      hidden, h0f[0], h0b[0], h1f[0], h1b[0]);

  // ---- recurrent loop: 2 fused launches per timestep ----
  const dim3 gGru(B_ / 16, H_ / 64);   // (4, 32)
  const dim3 bGru(32, 4);
  for (int t = 0; t < TS_; ++t) {
    const int pg = t & 1, np = pg ^ 1;
    // layer 0: x_t = target[:, t, :]
    gru_step<<<gGru, bGru, 0, stream>>>(
        xb + (long)t * D_, (long)T_ * D_, D_,
        wih0b, whh0b, bih0, bhh0,
        h0f[pg], h0b[pg], h0f[np], h0b[np], (/*ys*/ (__bf16*)nullptr), t);
    // layer 1: x = h0_new; also emits ys[:, t, :]
    gru_step<<<gGru, bGru, 0, stream>>>(
        h0b[np], (long)H_, H_,
        wih1b, whh1b, bih1, bhh1,
        h1f[pg], h1b[pg], h1f[np], h1b[np], ysb, t);
  }
  const int fin = TS_ & 1;  // final parity buffer index (255 steps -> 1)

  // ---- MLP head over all (B*(T-1)) rows ----
  const dim3 bG(32, 4);
  gemm_bias_act<<<dim3((int)(MROWS / 32), H_ / 256), bG, 0, stream>>>(
      ysb, w1b, b1, nullptr, m1b, H_, H_, /*relu=*/1);
  gemm_bias_act<<<dim3((int)(MROWS / 32), D_ / 256), bG, 0, stream>>>(
      m1b, w2b, b2, nullptr, m2b, H_, D_, /*relu=*/1);
  gemm_bias_act<<<dim3((int)(MROWS / 32), D_ / 256), bG, 0, stream>>>(
      m2b, w3b, b3, (float*)d_out, nullptr, D_, D_, /*relu=*/0);

  // ---- final hidden state: concat after (B,T-1,D) output ----
  float* outHid = (float*)d_out + MROWS * D_;
  copy_hidden_out<<<(int)((2 * SZ_HB + 255) / 256), 256, 0, stream>>>(
      h0f[fin], h1f[fin], outHid);
}
